// MultiHeadAttention_CF_29403346108539
// MI455X (gfx1250) — matrix operand order repaired and checked by reading, not compiled, not run on hardware
//
#include <hip/hip_runtime.h>

typedef __attribute__((ext_vector_type(16))) __bf16 v16bf;
typedef __attribute__((ext_vector_type(8)))  __bf16 v8bf;
typedef __attribute__((ext_vector_type(8)))  float  v8f;

#define D_MODEL  1024
#define N_HEAD   16
#define HEAD_DIM 64
#define BATCH    4
#define SEQ      2048
#define M_TOT    (BATCH * SEQ)   // 8192 tokens

union FragBF { v16bf v; v8bf h[2]; };

__device__ inline v8f wmma_bf16(v16bf a, v16bf b, v8f c) {
    return __builtin_amdgcn_wmma_f32_16x16x32_bf16(
        /*neg_a=*/false, a, /*neg_b=*/false, b,
        /*c_mod=*/(short)0, c, /*reuse_a=*/false, /*reuse_b=*/false);
}

// Async 16-byte global -> LDS copy (no VGPR round trip). Tracked by ASYNCcnt.
// ldsOff is the wave-relative LDS byte offset = low 32 bits of the generic
// pointer to a __shared__ object (flat->LDS aperture truncation, ISA 10.2).
__device__ inline void async_copy_b128(unsigned ldsOff, const __bf16* g) {
    asm volatile("global_load_async_to_lds_b128 %0, %1, off"
                 :: "v"(ldsOff), "v"((unsigned long long)(uintptr_t)g)
                 : "memory");
}
__device__ inline void wait_async0() {
    asm volatile("s_wait_asynccnt 0" ::: "memory");
}

// A-matrix 16x32 bf16 fragment: lane holds row (lane&15); lanes 0-15 carry
// K = 0..15, lanes 16-31 carry K = 16..31 (contiguous, matches fragB).
__device__ inline v16bf load_fragA(const __bf16* __restrict__ base, int row,
                                   int ldk, int kBase, int lane) {
    int r  = lane & 15;
    int k0 = kBase + ((lane & 16) ? 16 : 0);
    const __bf16* p = base + (size_t)(row + r) * ldk + k0;
    FragBF f;
    f.h[0] = *(const v8bf*)(p);
    f.h[1] = *(const v8bf*)(p + 8);
    return f.v;
}

// B-matrix 32x16 bf16 fragment: lane holds column (lane&15); lanes 0-15 carry
// K = 0..15, lanes 16-31 carry K = 16..31 (contiguous 32B per lane).
__device__ inline v16bf load_fragB(const __bf16* __restrict__ base, int col,
                                   int ldk, int kBase, int lane) {
    int c  = lane & 15;
    int k0 = kBase + ((lane & 16) ? 16 : 0);
    const __bf16* p = base + (size_t)(col + c) * ldk + k0;
    FragBF f;
    f.h[0] = *(const v8bf*)(p);
    f.h[1] = *(const v8bf*)(p + 8);
    return f.v;
}

// B-fragment from an LDS tile pointer (row already selected by caller).
__device__ inline v16bf lds_fragB(const __bf16* p) {
    FragBF f;
    f.h[0] = *(const v8bf*)(p);
    f.h[1] = *(const v8bf*)(p + 8);
    return f.v;
}

__global__ __launch_bounds__(256) void f32_to_bf16_kernel(
    const float* __restrict__ in, __bf16* __restrict__ out, int nElem) {
    int idx = blockIdx.x * 256 + threadIdx.x;
    if (idx < nElem) out[idx] = (__bf16)in[idx];
}

// C = A[M_TOT,1024] * W[1024,1024]^T + bias.  64x64 tile per wave,
// 8 waves (2M x 4N) per block -> 128(M) x 256(N) block tile.
// mode 0: head-split bf16 [B,H,S,Dh] (Q, K)
// mode 1: transposed bf16 [B,H,Dh,S] (V)
// mode 2: flat f32 [M_TOT, D_MODEL]  (final output)
__global__ __launch_bounds__(256) void gemm_bf16_kernel(
    const __bf16* __restrict__ A, const __bf16* __restrict__ W,
    const float* __restrict__ bias, __bf16* __restrict__ outb,
    float* __restrict__ outf, int mode) {
    int lane  = threadIdx.x & 31;
    int w     = threadIdx.x >> 5;
    int waveM = w & 1, waveN = w >> 1;
    int mBase = blockIdx.x * 128 + waveM * 64;
    int nBase = blockIdx.y * 256 + waveN * 64;

    v8f acc[4][4];
#pragma unroll
    for (int i = 0; i < 4; ++i)
#pragma unroll
        for (int j = 0; j < 4; ++j) acc[i][j] = (v8f){};

    for (int kb = 0; kb < D_MODEL; kb += 32) {
        v16bf a[4], b[4];
#pragma unroll
        for (int i = 0; i < 4; ++i)
            a[i] = load_fragA(A, mBase + 16 * i, D_MODEL, kb, lane);
#pragma unroll
        for (int j = 0; j < 4; ++j)
            b[j] = load_fragB(W, nBase + 16 * j, D_MODEL, kb, lane);
#pragma unroll
        for (int i = 0; i < 4; ++i)
#pragma unroll
            for (int j = 0; j < 4; ++j)
                acc[i][j] = wmma_bf16(a[i], b[j], acc[i][j]);
    }

    int n  = lane & 15;
    int mo = (lane & 16) ? 8 : 0;
#pragma unroll
    for (int j2 = 0; j2 < 4; ++j2) {
        int col  = nBase + j2 * 16 + n;
        float bv = bias[col];
#pragma unroll
        for (int i2 = 0; i2 < 4; ++i2) {
#pragma unroll
            for (int e = 0; e < 8; ++e) {
                int row   = mBase + i2 * 16 + mo + e;     // token 0..8191
                float val = acc[i2][j2][e] + bv;
                if (mode == 2) {
                    outf[(size_t)row * D_MODEL + col] = val;
                } else {
                    int b = row >> 11, s = row & 2047;
                    int h = col >> 6,  d = col & 63;
                    if (mode == 1)
                        outb[((size_t)(b * N_HEAD + h) * HEAD_DIM + d) * SEQ + s] = (__bf16)val;
                    else
                        outb[((size_t)(b * N_HEAD + h) * SEQ + s) * HEAD_DIM + d] = (__bf16)val;
                }
            }
        }
    }
}

// Flash attention. One block = 8 waves = 128 consecutive query rows of one
// (b,h). K/V tiles for each 32-key step are staged in LDS with async
// global->LDS copies (double-buffered, one barrier + s_wait_asynccnt per
// step). exp(P) goes through per-wave LDS to become an A-fragment.
__global__ __launch_bounds__(256) void attn_kernel(
    const __bf16* __restrict__ Q, const __bf16* __restrict__ K,
    const __bf16* __restrict__ Vt, __bf16* __restrict__ O) {
    __shared__ __align__(16) __bf16 Ktile[2][32][72];   // keys x d, padded
    __shared__ __align__(16) __bf16 Vtile[2][64][40];   // d x keys, padded
    __shared__ __align__(16) __bf16 pstage[8][16 * 40];

    int tid  = threadIdx.x;
    int lane = tid & 31;
    int w    = tid >> 5;
    int bh   = blockIdx.x >> 4;            // 16 blocks per (b,h)
    int qchunk = blockIdx.x & 15;
    int qbase  = qchunk * 128 + w * 16;

    const __bf16* Qh = Q  + (size_t)bh * SEQ * HEAD_DIM;
    const __bf16* Kh = K  + (size_t)bh * SEQ * HEAD_DIM;
    const __bf16* Vh = Vt + (size_t)bh * HEAD_DIM * SEQ;

    // cooperative tile fill: 256 threads, one async 16B chunk each per tile
    int krow = tid >> 3, kc8 = (tid & 7) * 8;   // 32 x 64
    int vrow = tid >> 2, vc8 = (tid & 3) * 8;   // 64 x 32
    auto fill = [&](int buf, int kb) {
        async_copy_b128((unsigned)(uintptr_t)&Ktile[buf][krow][kc8],
                        Kh + (size_t)(kb + krow) * HEAD_DIM + kc8);
        async_copy_b128((unsigned)(uintptr_t)&Vtile[buf][vrow][vc8],
                        Vh + (size_t)vrow * SEQ + kb + vc8);
    };
    fill(0, 0);

    v16bf qa0 = load_fragA(Qh, qbase, HEAD_DIM, 0,  lane);
    v16bf qa1 = load_fragA(Qh, qbase, HEAD_DIM, 32, lane);

    v8f o0 = {}, o1 = {}, o2 = {}, o3 = {};
    float m_s[8], l_s[8];
#pragma unroll
    for (int i = 0; i < 8; ++i) { m_s[i] = -__builtin_inff(); l_s[i] = 0.f; }

    int n    = lane & 15;
    int mo   = (lane & 16) ? 8 : 0;
    int c    = lane & 15;                   // fragment column lane
    int koff = (lane & 16) ? 16 : 0;        // fragB K half select

    for (int kb = 0; kb < SEQ; kb += 32) {
        int buf = (kb >> 5) & 1;
        wait_async0();                      // my async fill of tile[buf] landed
        __syncthreads();                    // everyone's landed; other buf free
        if (kb + 32 < SEQ) fill(buf ^ 1, kb + 32);

        // scores: two 16-key subtiles, k = Dh = 64 -> 2 WMMA steps each
        v16bf kf00 = lds_fragB(&Ktile[buf][c][koff]);
        v16bf kf01 = lds_fragB(&Ktile[buf][c][32 + koff]);
        v16bf kf10 = lds_fragB(&Ktile[buf][16 + c][koff]);
        v16bf kf11 = lds_fragB(&Ktile[buf][16 + c][32 + koff]);
        v8f s0 = {}, s1 = {};
        s0 = wmma_bf16(qa0, kf00, s0);
        s0 = wmma_bf16(qa1, kf01, s0);
        s1 = wmma_bf16(qa0, kf10, s1);
        s1 = wmma_bf16(qa1, kf11, s1);

#pragma unroll
        for (int i = 0; i < 8; ++i) {
            float e0 = s0[i] * 0.125f;           // 1/sqrt(64)
            float e1 = s1[i] * 0.125f;
            float rm = fmaxf(e0, e1);
#pragma unroll
            for (int off = 8; off >= 1; off >>= 1)
                rm = fmaxf(rm, __shfl_xor(rm, off, 16));
            float mnew  = fmaxf(m_s[i], rm);
            float alpha = __expf(m_s[i] - mnew);
            float p0    = __expf(e0 - mnew);
            float p1    = __expf(e1 - mnew);
            float rs    = p0 + p1;
#pragma unroll
            for (int off = 8; off >= 1; off >>= 1)
                rs += __shfl_xor(rs, off, 16);
            l_s[i] = l_s[i] * alpha + rs;
            m_s[i] = mnew;
            o0[i] = o0[i] * alpha;
            o1[i] = o1[i] * alpha;
            o2[i] = o2[i] * alpha;
            o3[i] = o3[i] * alpha;
            pstage[w][(mo + i) * 40 + n]      = (__bf16)p0;
            pstage[w][(mo + i) * 40 + 16 + n] = (__bf16)p1;
        }

        // re-read exp(P) (16x32) as an A-layout fragment (LDS in-order per wave)
        int k0 = (lane & 16) ? 16 : 0;
        FragBF pf;
        pf.h[0] = *(const v8bf*)&pstage[w][c * 40 + k0];
        pf.h[1] = *(const v8bf*)&pstage[w][c * 40 + k0 + 8];

        v16bf vf0 = lds_fragB(&Vtile[buf][ 0 + c][koff]);
        v16bf vf1 = lds_fragB(&Vtile[buf][16 + c][koff]);
        v16bf vf2 = lds_fragB(&Vtile[buf][32 + c][koff]);
        v16bf vf3 = lds_fragB(&Vtile[buf][48 + c][koff]);
        o0 = wmma_bf16(pf.v, vf0, o0);
        o1 = wmma_bf16(pf.v, vf1, o1);
        o2 = wmma_bf16(pf.v, vf2, o2);
        o3 = wmma_bf16(pf.v, vf3, o3);
    }

    // normalize and write merged-head bf16 [B,S,D]
    int h = bh & (N_HEAD - 1);
    size_t tokBase = (size_t)(bh >> 4) * SEQ + qbase + mo;
#pragma unroll
    for (int i = 0; i < 8; ++i) {
        float inv = 1.0f / l_s[i];
        size_t ro = (tokBase + i) * D_MODEL + (size_t)h * HEAD_DIM;
        O[ro +  0 + n] = (__bf16)(o0[i] * inv);
        O[ro + 16 + n] = (__bf16)(o1[i] * inv);
        O[ro + 32 + n] = (__bf16)(o2[i] * inv);
        O[ro + 48 + n] = (__bf16)(o3[i] * inv);
    }
}

extern "C" void kernel_launch(void* const* d_in, const int* in_sizes, int n_in,
                              void* d_out, int out_size, void* d_ws, size_t ws_size,
                              hipStream_t stream) {
    const float* x  = (const float*)d_in[0];
    const float* Wq = (const float*)d_in[1];
    const float* bq = (const float*)d_in[2];
    const float* Wk = (const float*)d_in[3];
    const float* bk = (const float*)d_in[4];
    const float* Wv = (const float*)d_in[5];
    const float* bv = (const float*)d_in[6];
    const float* Wo = (const float*)d_in[7];
    const float* bo = (const float*)d_in[8];

    char* ws = (char*)d_ws;
    size_t o = 0;
    const size_t xbytes = (size_t)M_TOT * D_MODEL * 2;     // 16 MB
    const size_t wbytes = (size_t)D_MODEL * D_MODEL * 2;   // 2 MB
    __bf16* xbf = (__bf16*)(ws + o); o += xbytes;
    __bf16* wqb = (__bf16*)(ws + o); o += wbytes;
    __bf16* wkb = (__bf16*)(ws + o); o += wbytes;
    __bf16* wvb = (__bf16*)(ws + o); o += wbytes;
    __bf16* wob = (__bf16*)(ws + o); o += wbytes;
    __bf16* Qb  = (__bf16*)(ws + o); o += xbytes;
    __bf16* Kb  = (__bf16*)(ws + o); o += xbytes;
    __bf16* Vtb = (__bf16*)(ws + o); o += xbytes;
    __bf16* Ab  = (__bf16*)(ws + o); o += xbytes;

    const int nX = M_TOT * D_MODEL;      // 8,388,608
    const int nW = D_MODEL * D_MODEL;    // 1,048,576
    f32_to_bf16_kernel<<<nX / 256, 256, 0, stream>>>(x,  xbf, nX);
    f32_to_bf16_kernel<<<nW / 256, 256, 0, stream>>>(Wq, wqb, nW);
    f32_to_bf16_kernel<<<nW / 256, 256, 0, stream>>>(Wk, wkb, nW);
    f32_to_bf16_kernel<<<nW / 256, 256, 0, stream>>>(Wv, wvb, nW);
    f32_to_bf16_kernel<<<nW / 256, 256, 0, stream>>>(Wo, wob, nW);

    dim3 gg(M_TOT / 128, D_MODEL / 256);
    gemm_bf16_kernel<<<gg, 256, 0, stream>>>(xbf, wqb, bq, Qb,  nullptr, 0);
    gemm_bf16_kernel<<<gg, 256, 0, stream>>>(xbf, wkb, bk, Kb,  nullptr, 0);
    gemm_bf16_kernel<<<gg, 256, 0, stream>>>(xbf, wvb, bv, Vtb, nullptr, 1);

    // 1024 blocks: 16 blocks of 128 query rows per (b,h)
    attn_kernel<<<BATCH * N_HEAD * (SEQ / 128), 256, 0, stream>>>(Qb, Kb, Vtb, Ab);

    gemm_bf16_kernel<<<gg, 256, 0, stream>>>(Ab, wob, bo, nullptr, (float*)d_out, 2);
}
